// MultilayerGRU_37830071943928
// MI455X (gfx1250) — compile-verified, matrix operand order checked by reading, not run on hardware
//
#include <hip/hip_runtime.h>
#include <hip/hip_bf16.h>

// ---------------- problem constants ----------------
#define S_LEN   1024
#define BATCH   32
#define HDIM    512
#define IDIM    128
#define ODIM    128
#define NLAYER  3
#define NCG     8        // column-groups per layer; each owns 64 of the 512 features

typedef __bf16 bf16;
typedef __attribute__((ext_vector_type(16))) bf16  v16bf;
typedef __attribute__((ext_vector_type(8)))  bf16  v8bf;
typedef __attribute__((ext_vector_type(8)))  float v8f;

// ---------------- workspace layout (bytes), total ~12.5 MB ----------------
constexpr size_t OFF_CNT  = 0;                                     // 16 ints (cntA[3], cntB[3])
constexpr size_t OFF_W0   = 256;                                   // 3 x (512x128) bf16, gates z,r,g
constexpr size_t SZ_W0    = (size_t)HDIM * IDIM * 2;
constexpr size_t OFF_WX   = OFF_W0 + 3 * SZ_W0;                    // 3 x (512x512) bf16
constexpr size_t SZ_WH    = (size_t)HDIM * HDIM * 2;
constexpr size_t OFF_WH   = OFF_WX + 3 * SZ_WH;                    // 3 x (512x512) bf16
constexpr size_t OFF_WOUT = OFF_WH + 3 * SZ_WH;                    // (128x512) bf16
constexpr size_t OFF_XT   = OFF_WOUT + (size_t)ODIM * HDIM * 2;    // (S,B,I) bf16
constexpr size_t OFF_H    = OFF_XT + (size_t)S_LEN * BATCH * IDIM * 2; // 3 layers x 2 slots x (B,H) bf16
constexpr size_t OFF_RH   = OFF_H + (size_t)NLAYER * 2 * BATCH * HDIM * 2; // 3 x (B,H) bf16
constexpr size_t OFF_H32  = OFF_RH + (size_t)NLAYER * BATCH * HDIM * 2;    // 3 x (B,H) f32
constexpr size_t WS_NEED  = OFF_H32 + (size_t)NLAYER * BATCH * HDIM * 4;

constexpr size_t OUT_HID  = (size_t)BATCH * S_LEN * ODIM;          // offset of hidden_state in d_out

// ---------------- fragment load (bf16, 16x32 A / 32x16 B symmetric layout) ----------------
// A (16-bit, 16x32): lane L: row = (L&15); hi = L>>4;
//   halves 0..7  = K[hi*8 .. hi*8+7],  halves 8..15 = K[16+hi*8 .. 16+hi*8+7]
// Weights stored (N,K) row-major load identically with row = output feature.
__device__ __forceinline__ v16bf load_frag(const bf16* __restrict__ base, int rowBase,
                                           int ld, int k0, int lane) {
    int m  = lane & 15;
    int hi = lane >> 4;
    const bf16* p = base + (size_t)(rowBase + m) * ld + k0 + hi * 8;
    v8bf lo = *(const v8bf*)(p);
    v8bf hv = *(const v8bf*)(p + 16);
    v16bf r;
#pragma unroll
    for (int i = 0; i < 8; ++i) { r[i] = lo[i]; r[i + 8] = hv[i]; }
    return r;
}

__device__ __forceinline__ v8f wmma_acc(v8f acc, const bf16* __restrict__ A, int ldA, int mBase,
                                        const bf16* __restrict__ W, int nBase, int K, int lane) {
    for (int k0 = 0; k0 < K; k0 += 32) {
        v16bf a = load_frag(A, mBase, ldA, k0, lane);
        v16bf b = load_frag(W, nBase, K, k0, lane);
        acc = __builtin_amdgcn_wmma_f32_16x16x32_bf16(false, a, false, b, (short)0, acc,
                                                      false, false);
    }
    return acc;
}

__device__ __forceinline__ void spin_until(int* p, int target) {
    while (__hip_atomic_load(p, __ATOMIC_RELAXED, __HIP_MEMORY_SCOPE_AGENT) < target)
        __builtin_amdgcn_s_sleep(1);
}

__device__ __forceinline__ float sigmoidf_(float x) { return 1.0f / (1.0f + __expf(-x)); }

// ---------------- prep kernels ----------------
__global__ void f2bf_kernel(const float* __restrict__ s, bf16* __restrict__ d, int n) {
    int i = blockIdx.x * blockDim.x + threadIdx.x;
    if (i < n) d[i] = (bf16)s[i];
}

__global__ void xpose_kernel(const float* __restrict__ x, bf16* __restrict__ xT) {
    int i = blockIdx.x * blockDim.x + threadIdx.x;   // flat over (S,B,I)
    int s = i / (BATCH * IDIM);
    int r = i % (BATCH * IDIM);
    int b = r / IDIM;
    int k = r % IDIM;
    xT[i] = (bf16)x[((size_t)b * S_LEN + s) * IDIM + k];
}

__global__ void init_kernel(const float* __restrict__ h0, char* __restrict__ ws) {
    int i = blockIdx.x * blockDim.x + threadIdx.x;   // flat over (L,B,H) = 49152
    if (i < 16) ((int*)(ws + OFF_CNT))[i] = 0;
    int j = i / (BATCH * HDIM);
    int r = i % (BATCH * HDIM);
    int b = r / HDIM;
    int n = r % HDIM;
    float v = h0[(size_t)b * NLAYER * HDIM + j * HDIM + n];
    ((float*)(ws + OFF_H32))[j * (BATCH * HDIM) + r] = v;
    // slot 0 of layer j holds h_{-1} (t=0 reads slot t&1 == 0)
    ((bf16*)(ws + OFF_H))[(j * 2 + 0) * (BATCH * HDIM) + r] = (bf16)v;
}

// ---------------- persistent recurrent kernel ----------------
// grid = NLAYER*NCG blocks, 256 threads (8 wave32s).
// Block (j,c): layer j, output-feature slice [c*64, c*64+64).
// Wave w: M-tile mt=w&1 (batch rows), N-tile nt=w>>1 within the slice.
__global__ void __launch_bounds__(256, 1)
gru_recurrent(const float* __restrict__ b0z, const float* __restrict__ b0r,
              const float* __restrict__ b0g, const float* __restrict__ bxz,
              const float* __restrict__ bxr, const float* __restrict__ bxg,
              const float* __restrict__ bout, char* __restrict__ ws,
              float* __restrict__ out) {
    const int j   = blockIdx.x / NCG;
    const int c   = blockIdx.x % NCG;
    const int tid = threadIdx.x;
    const int lane = tid & 31;
    const int wave = tid >> 5;
    const int lane15 = lane & 15;
    const int hi = lane >> 4;
    const int mt = wave & 1;
    const int nt = wave >> 1;
    const int mBase = mt * 16;
    const int nAbs  = c * 64 + nt * 16;

    int*  cntA = (int*)(ws + OFF_CNT);
    int*  cntB = cntA + NLAYER;
    const bf16* W0   = (const bf16*)(ws + OFF_W0);
    const bf16* WX   = (const bf16*)(ws + OFF_WX);
    const bf16* WH   = (const bf16*)(ws + OFF_WH);
    const bf16* WOUTb= (const bf16*)(ws + OFF_WOUT);
    const bf16* XT   = (const bf16*)(ws + OFF_XT);
    bf16*  HSL  = (bf16*)(ws + OFF_H);
    bf16*  RH   = (bf16*)(ws + OFF_RH);
    float* H32  = (float*)(ws + OFF_H32);

    const bf16* Wz  = (j == 0) ? W0 + 0 * (HDIM * IDIM) : WX + 0 * (HDIM * HDIM);
    const bf16* Wr  = (j == 0) ? W0 + 1 * (HDIM * IDIM) : WX + 1 * (HDIM * HDIM);
    const bf16* Wg  = (j == 0) ? W0 + 2 * (HDIM * IDIM) : WX + 2 * (HDIM * HDIM);
    const bf16* Whz = WH + 0 * (HDIM * HDIM);
    const bf16* Whr = WH + 1 * (HDIM * HDIM);
    const bf16* Whg = WH + 2 * (HDIM * HDIM);
    const float* bzp = (j == 0) ? b0z : bxz;
    const float* brp = (j == 0) ? b0r : bxr;
    const float* bgp = (j == 0) ? b0g : bxg;
    const int Kx = (j == 0) ? IDIM : HDIM;

    bf16*  rh_l  = RH  + j * (BATCH * HDIM);
    float* h32_l = H32 + j * (BATCH * HDIM);

    const float bz = bzp[nAbs + lane15];
    const float br = brp[nAbs + lane15];
    const float bg = bgp[nAbs + lane15];

    for (int t = 0; t < S_LEN; ++t) {
        // ---------- phase 1: z, r ----------
        if (tid == 0) {
            spin_until(&cntB[j], t * NCG);                 // own h_{t-1} + rh buffer free
            if (j > 0) spin_until(&cntB[j - 1], (t + 1) * NCG); // input h^{j-1}_t ready
        }
        __syncthreads();
        __threadfence();

        const bf16* hprev = HSL + ((j * 2 + (t & 1)) * (BATCH * HDIM));
        const bf16* xin   = (j == 0) ? (XT + (size_t)t * BATCH * IDIM)
                                     : (HSL + (((j - 1) * 2 + ((t + 1) & 1)) * (BATCH * HDIM)));

        v8f accZ = {bz, bz, bz, bz, bz, bz, bz, bz};
        accZ = wmma_acc(accZ, xin, Kx, mBase, Wz, nAbs, Kx, lane);
        accZ = wmma_acc(accZ, hprev, HDIM, mBase, Whz, nAbs, HDIM, lane);

        v8f accR = {br, br, br, br, br, br, br, br};
        accR = wmma_acc(accR, xin, Kx, mBase, Wr, nAbs, Kx, lane);
        accR = wmma_acc(accR, hprev, HDIM, mBase, Whr, nAbs, HDIM, lane);

        float zreg[8];
#pragma unroll
        for (int v = 0; v < 8; ++v) {
            int b   = mBase + v + 8 * hi;
            int idx = b * HDIM + nAbs + lane15;
            zreg[v]   = sigmoidf_(accZ[v]);
            float r   = sigmoidf_(accR[v]);
            rh_l[idx] = (bf16)(r * h32_l[idx]);
        }
        __threadfence();
        __syncthreads();
        if (tid == 0)
            __hip_atomic_fetch_add(&cntA[j], 1, __ATOMIC_RELAXED, __HIP_MEMORY_SCOPE_AGENT);

        // ---------- phase 2: g, h_new ----------
        if (tid == 0) {
            spin_until(&cntA[j], (t + 1) * NCG);           // full r*h ready
            if (j < NLAYER - 1) spin_until(&cntB[j + 1], (t - 1) * NCG); // h slot consumed
        }
        __syncthreads();
        __threadfence();

        v8f accG = {bg, bg, bg, bg, bg, bg, bg, bg};
        accG = wmma_acc(accG, xin, Kx, mBase, Wg, nAbs, Kx, lane);
        accG = wmma_acc(accG, rh_l, HDIM, mBase, Whg, nAbs, HDIM, lane);

        bf16* hnew = HSL + ((j * 2 + ((t + 1) & 1)) * (BATCH * HDIM));
#pragma unroll
        for (int v = 0; v < 8; ++v) {
            int b    = mBase + v + 8 * hi;
            int col  = nAbs + lane15;
            int idx  = b * HDIM + col;
            float g  = tanhf(accG[v]);
            float ho = h32_l[idx];
            float hn = zreg[v] * ho + (1.0f - zreg[v]) * g;
            h32_l[idx] = hn;
            hnew[idx]  = (bf16)hn;
            if (t == S_LEN - 1)
                out[OUT_HID + (size_t)b * (NLAYER * HDIM) + j * HDIM + col] = hn;
        }
        __threadfence();
        __syncthreads();
        if (tid == 0)
            __hip_atomic_fetch_add(&cntB[j], 1, __ATOMIC_RELAXED, __HIP_MEMORY_SCOPE_AGENT);

        // ---------- phase 3 (last layer only): out_t = h2_t @ Wout^T + bout ----------
        if (j == NLAYER - 1) {
            if (tid == 0) spin_until(&cntB[j], (t + 1) * NCG);   // full h2_t ready
            __syncthreads();
            __threadfence();
            if (wave < 2) {                                      // 2 M-tiles x 1 N-tile per group
                const bf16* h2 = HSL + ((j * 2 + ((t + 1) & 1)) * (BATCH * HDIM));
                int mB = wave * 16;
                int oB = c * 16;
                float bo = bout[oB + lane15];
                v8f acc = {bo, bo, bo, bo, bo, bo, bo, bo};
                acc = wmma_acc(acc, h2, HDIM, mB, WOUTb, oB, HDIM, lane);
#pragma unroll
                for (int v = 0; v < 8; ++v) {
                    int b = mB + v + 8 * hi;
                    out[(size_t)b * S_LEN * ODIM + (size_t)t * ODIM + oB + lane15] = acc[v];
                }
            }
            __syncthreads();
        }
    }
}

// ---------------- host launcher ----------------
extern "C" void kernel_launch(void* const* d_in, const int* in_sizes, int n_in,
                              void* d_out, int out_size, void* d_ws, size_t ws_size,
                              hipStream_t stream) {
    (void)in_sizes; (void)n_in; (void)out_size; (void)ws_size;
    const float* x    = (const float*)d_in[0];
    const float* h0   = (const float*)d_in[1];
    const float* W0z  = (const float*)d_in[2];  const float* b0z = (const float*)d_in[3];
    const float* W0r  = (const float*)d_in[4];  const float* b0r = (const float*)d_in[5];
    const float* W0g  = (const float*)d_in[6];  const float* b0g = (const float*)d_in[7];
    const float* Wxz  = (const float*)d_in[8];  const float* bxz = (const float*)d_in[9];
    const float* Wxr  = (const float*)d_in[10]; const float* bxr = (const float*)d_in[11];
    const float* Wxg  = (const float*)d_in[12]; const float* bxg = (const float*)d_in[13];
    const float* Whz  = (const float*)d_in[14];
    const float* Whr  = (const float*)d_in[15];
    const float* Whg  = (const float*)d_in[16];
    const float* Wout = (const float*)d_in[17]; const float* bout = (const float*)d_in[18];

    char*  ws  = (char*)d_ws;
    float* out = (float*)d_out;

    auto cvt = [&](const float* src, size_t off, int n) {
        f2bf_kernel<<<(n + 255) / 256, 256, 0, stream>>>(src, (bf16*)(ws + off), n);
    };
    // weights -> bf16 (gate order z, r, g)
    cvt(W0z, OFF_W0 + 0 * SZ_W0, HDIM * IDIM);
    cvt(W0r, OFF_W0 + 1 * SZ_W0, HDIM * IDIM);
    cvt(W0g, OFF_W0 + 2 * SZ_W0, HDIM * IDIM);
    cvt(Wxz, OFF_WX + 0 * SZ_WH, HDIM * HDIM);
    cvt(Wxr, OFF_WX + 1 * SZ_WH, HDIM * HDIM);
    cvt(Wxg, OFF_WX + 2 * SZ_WH, HDIM * HDIM);
    cvt(Whz, OFF_WH + 0 * SZ_WH, HDIM * HDIM);
    cvt(Whr, OFF_WH + 1 * SZ_WH, HDIM * HDIM);
    cvt(Whg, OFF_WH + 2 * SZ_WH, HDIM * HDIM);
    cvt(Wout, OFF_WOUT, ODIM * HDIM);

    // x (B,S,I) fp32 -> xT (S,B,I) bf16
    {
        int n = S_LEN * BATCH * IDIM;
        xpose_kernel<<<(n + 255) / 256, 256, 0, stream>>>(x, (bf16*)(ws + OFF_XT));
    }
    // h0 -> fp32 state + bf16 slot0, zero sync counters
    {
        int n = NLAYER * BATCH * HDIM;
        init_kernel<<<(n + 255) / 256, 256, 0, stream>>>(h0, ws);
    }

    // persistent wavefront-pipelined GRU: 3 layers x 8 column groups
    gru_recurrent<<<dim3(NLAYER * NCG), dim3(256), 0, stream>>>(
        b0z, b0r, b0g, bxz, bxr, bxg, bout, ws, out);
}
